// MaxSigmoidAttnFusionBLock_11269994185005
// MI455X (gfx1250) — matrix operand order repaired and checked by reading, not compiled, not run on hardware
//
#include <hip/hip_runtime.h>
#include <hip/hip_bf16.h>
#include <math.h>

// ---------------------------------------------------------------------------
// MaxSigmoidAttnFusionBlock for MI455X (gfx1250)
//   - proj / attn GEMMs: exact fp32 via V_WMMA_F32_16X16X4_F32
//   - 3x3 conv (dominant 38.7 GFLOP): bf16x3 split via V_WMMA_F32_16X16X32_BF16
// B=8, C=256, H=W=64 (HW=4096), L=80, D=256, heads=8, hc=32
// ---------------------------------------------------------------------------

typedef __attribute__((ext_vector_type(2)))  float        v2f;
typedef __attribute__((ext_vector_type(8)))  float        v8f;
typedef __attribute__((ext_vector_type(16))) __bf16       v16bf;
typedef __attribute__((ext_vector_type(4)))  unsigned int v4u;
typedef __attribute__((ext_vector_type(8)))  unsigned int v8u;

#define NB 8
#define NC 256
#define NHW 4096
#define NL 80
#define NHEADS 8
#define NHC 32
#define KCONV 2304   // 256*9

__device__ __forceinline__ v8f wmma4(v2f a, v2f b, v8f c) {
    return __builtin_amdgcn_wmma_f32_16x16x4_f32(
        false, a, false, b, (short)0, c, false, false);
}
__device__ __forceinline__ v8f wmma32(v16bf a, v16bf b, v8f c) {
    return __builtin_amdgcn_wmma_f32_16x16x32_bf16(
        false, a, false, b, (short)0, c, false, false);
}

// round-to-nearest-even fp32 -> bf16 bits
__device__ __forceinline__ unsigned bf16r(float x) {
    unsigned u = __float_as_uint(x);
    return (u + 0x7fffu + ((u >> 16) & 1u)) >> 16;
}
// split two fp32 into packed (hi,lo) bf16 pairs: even k -> [15:0], odd -> [31:16]
__device__ __forceinline__ void splitpack(float x0, float x1,
                                          unsigned& hi, unsigned& lo) {
    unsigned h0 = bf16r(x0), h1 = bf16r(x1);
    float r0 = x0 - __uint_as_float(h0 << 16);
    float r1 = x1 - __uint_as_float(h1 << 16);
    hi = h0 | (h1 << 16);
    lo = bf16r(r0) | (bf16r(r1) << 16);
}
// build v16bf fragment from two 16B LDS chunks at p and p+d2 dwords
__device__ __forceinline__ v16bf ldfrag(const unsigned* p, int d2) {
    union { v8u u; v16bf b; } f;
    v4u x = *(const v4u*)p;
    v4u y = *(const v4u*)(p + d2);
    f.u[0] = x[0]; f.u[1] = x[1]; f.u[2] = x[2]; f.u[3] = x[3];
    f.u[4] = y[0]; f.u[5] = y[1]; f.u[6] = y[2]; f.u[7] = y[3];
    return f.b;
}

// ---------------------------------------------------------------------------
// Kernel 1: i_feat[b][d][hw] = sum_c proj_w[d][c]*img[b][c][hw] + proj_b[d]
// grid (64, 4, 8), block 128 (4 waves), wave = 16M x 64N, fp32 WMMA
// ---------------------------------------------------------------------------
__global__ __launch_bounds__(128) void proj_gemm(
    const float* __restrict__ img, const float* __restrict__ W,
    const float* __restrict__ bias, float* __restrict__ ifeat) {
    const int lane = threadIdx.x & 31;
    const int wv   = threadIdx.x >> 5;
    const int hi   = (lane >> 4) & 1;
    const int l16  = lane & 15;
    const int b    = blockIdx.z;
    const int m0   = blockIdx.y * 64 + wv * 16;
    const int n0   = blockIdx.x * 64;

    const float* Bp = img + (size_t)b * NC * NHW;
    const int arow = m0 + l16;
    const int n    = n0 + l16;

    v8f acc0 = {}, acc1 = {}, acc2 = {}, acc3 = {};

#pragma unroll 4
    for (int k = 0; k < NC; k += 4) {
        const int ka = k + 2 * hi;
        v2f a;
        a.x = W[arow * NC + ka];
        a.y = W[arow * NC + ka + 1];
        const float* bp = Bp + (size_t)ka * NHW;
        v2f b0 = { bp[n],      bp[NHW + n] };
        v2f b1 = { bp[n + 16], bp[NHW + n + 16] };
        v2f b2 = { bp[n + 32], bp[NHW + n + 32] };
        v2f b3 = { bp[n + 48], bp[NHW + n + 48] };
        acc0 = wmma4(a, b0, acc0);
        acc1 = wmma4(a, b1, acc1);
        acc2 = wmma4(a, b2, acc2);
        acc3 = wmma4(a, b3, acc3);
    }

#pragma unroll
    for (int r = 0; r < 8; ++r) {
        const int row = m0 + r + 8 * hi;
        const float bv = bias[row];
        float* op = ifeat + ((size_t)b * NC + row) * NHW + n;
        op[0]  = acc0[r] + bv;
        op[16] = acc1[r] + bv;
        op[32] = acc2[r] + bv;
        op[48] = acc3[r] + bv;
    }
}

// ---------------------------------------------------------------------------
// Kernel 2: gate[b][head][hw] = sigmoid(max_n attn /sqrt(32)+bias)*scale
// grid (64, 8, 8), block 128 (4 waves), wave = 16M x 80N, fp32 WMMA
// ---------------------------------------------------------------------------
__global__ __launch_bounds__(128) void attn_gate(
    const float* __restrict__ ifeat, const float* __restrict__ text,
    const float* __restrict__ attn_bias, const float* __restrict__ scale,
    float* __restrict__ gate) {
    const int lane = threadIdx.x & 31;
    const int wv   = threadIdx.x >> 5;
    const int hi   = (lane >> 4) & 1;
    const int l16  = lane & 15;
    const int b    = blockIdx.z;
    const int head = blockIdx.y;
    const int m0   = (blockIdx.x * 4 + wv) * 16;
    const int h0   = head * NHC;

    const float* Ap = ifeat + ((size_t)b * NC + h0) * NHW;  // [c][hw]
    const int m = m0 + l16;

    v8f acc[5] = {};

#pragma unroll
    for (int k = 0; k < NHC; k += 4) {
        const int ka = k + 2 * hi;
        v2f a = { Ap[(size_t)ka * NHW + m], Ap[(size_t)(ka + 1) * NHW + m] };
#pragma unroll
        for (int j = 0; j < 5; ++j) {
            const int n = j * 16 + l16;
            const float* tp = text + ((size_t)b * NL + n) * NC + h0 + ka;
            v2f bb = { tp[0], tp[1] };
            acc[j] = wmma4(a, bb, acc[j]);
        }
    }

    const float inv_sqrt_hc = 0.17677669529663687f;
    const float ab = attn_bias[head];
    const float sc = scale[head];

#pragma unroll
    for (int r = 0; r < 8; ++r) {
        float mx = acc[0][r];
#pragma unroll
        for (int j = 1; j < 5; ++j) mx = fmaxf(mx, acc[j][r]);
#pragma unroll
        for (int off = 1; off < 16; off <<= 1)
            mx = fmaxf(mx, __shfl_xor(mx, off, 32));
        if (l16 == 0) {
            const int row = m0 + r + 8 * hi;
            const float x = mx * inv_sqrt_hc + ab;
            gate[((size_t)b * NHEADS + head) * NHW + row] =
                sc / (1.0f + __expf(-x));
        }
    }
}

// ---------------------------------------------------------------------------
// Kernel 3: 3x3 conv + BN + gate, bf16x3 split on V_WMMA_F32_16X16X32_BF16.
// Implicit GEMM M=co(256) N=pixel K=2304. Block (256 thr, 8 waves) computes
// out[b][co0..+63][h][0..63]. Per K-chunk of 96: stage split/packed im2col
// operands in LDS ([row][48 packed-pair dwords], stride 52 -> b128-clean).
// grid (H=64, C/64=4, B=8)
// ---------------------------------------------------------------------------
#define KC   96   // K per LDS chunk
#define KD2  48   // packed-pair dwords per row per chunk
#define RS   52   // LDS row stride in dwords (16B aligned, conflict-free)

__global__ __launch_bounds__(256) void conv_bn_gate(
    const float* __restrict__ img, const float* __restrict__ cw,
    const float* __restrict__ gamma, const float* __restrict__ beta,
    const float* __restrict__ mean, const float* __restrict__ var,
    const float* __restrict__ gate, float* __restrict__ out) {
    __shared__ __align__(16) unsigned sAh[64 * RS];  // weight hi
    __shared__ __align__(16) unsigned sAl[64 * RS];  // weight lo
    __shared__ __align__(16) unsigned sBh[64 * RS];  // im2col hi
    __shared__ __align__(16) unsigned sBl[64 * RS];  // im2col lo

    const int tid  = threadIdx.x;
    const int lane = tid & 31;
    const int wv   = tid >> 5;
    const int hi   = (lane >> 4) & 1;
    const int l16  = lane & 15;
    const int h    = blockIdx.x;
    const int co0  = blockIdx.y * 64;
    const int b    = blockIdx.z;

    const int mloc = (wv & 3) * 16;         // wave M subtile within 64
    const int wpix = (wv >> 2) * 32 + l16;  // acc0 pixel col; acc1 = +16
    const int aoff = (mloc + l16) * RS + 4 * hi;   // A frag base (dwords)
    const int boff0 = wpix * RS + 4 * hi;          // B frag base acc0
    const int boff1 = (wpix + 16) * RS + 4 * hi;   // B frag base acc1

    v8f acc0 = {}, acc1 = {};

    for (int k0 = 0; k0 < KCONV; k0 += KC) {
        __syncthreads();
        // ---- stage split weights: rows = co_local, cols = packed K pairs
        for (int i = tid; i < 64 * KD2; i += 256) {
            const int row = i / KD2;
            const int c2  = i - row * KD2;
            const float* wp = cw + (size_t)(co0 + row) * KCONV + k0 + 2 * c2;
            unsigned hbits, lbits;
            splitpack(wp[0], wp[1], hbits, lbits);
            sAh[row * RS + c2] = hbits;
            sAl[row * RS + c2] = lbits;
        }
        // ---- stage split im2col image: rows = pixel, cols = packed K pairs
        for (int i = tid; i < 64 * KD2; i += 256) {
            const int p  = i & 63;     // pixel col (coalesced across tid)
            const int c2 = i >> 6;
            float x0 = 0.0f, x1 = 0.0f;
#pragma unroll
            for (int e = 0; e < 2; ++e) {
                const int k  = k0 + 2 * c2 + e;
                const int ci = k / 9;
                const int r9 = k - ci * 9;
                const int kh = r9 / 3;
                const int kw = r9 - kh * 3;
                const int hh = h + kh - 1;
                const int ww = p + kw - 1;
                float v = 0.0f;
                if (hh >= 0 && hh < 64 && ww >= 0 && ww < 64)
                    v = img[(((size_t)b * NC + ci) * 64 + hh) * 64 + ww];
                if (e == 0) x0 = v; else x1 = v;
            }
            unsigned hbits, lbits;
            splitpack(x0, x1, hbits, lbits);
            sBh[p * RS + c2] = hbits;
            sBl[p * RS + c2] = lbits;
        }
        __syncthreads();

        // ---- 3 x (K=32) steps: 3-term bf16 split, 2 N-subtiles
#pragma unroll
        for (int s = 0; s < 3; ++s) {
            const int sb = s * 16;
            v16bf ah = ldfrag(sAh + aoff + sb, 8);
            v16bf al = ldfrag(sAl + aoff + sb, 8);
            v16bf bh0 = ldfrag(sBh + boff0 + sb, 8);
            v16bf bl0 = ldfrag(sBl + boff0 + sb, 8);
            v16bf bh1 = ldfrag(sBh + boff1 + sb, 8);
            v16bf bl1 = ldfrag(sBl + boff1 + sb, 8);
            acc0 = wmma32(ah, bh0, acc0);
            acc0 = wmma32(ah, bl0, acc0);
            acc0 = wmma32(al, bh0, acc0);
            acc1 = wmma32(ah, bh1, acc1);
            acc1 = wmma32(ah, bl1, acc1);
            acc1 = wmma32(al, bh1, acc1);
        }
    }

    // ---- BN + per-head sigmoid gate epilogue
#pragma unroll
    for (int r = 0; r < 8; ++r) {
        const int co = co0 + mloc + r + 8 * hi;
        const float iv = gamma[co] * rsqrtf(var[co] + 1e-3f);
        const float ad = beta[co] - mean[co] * iv;
        const int p = h * 64 + wpix;
        const float* gp = gate + ((size_t)b * NHEADS + (co >> 5)) * NHW;
        const float g0 = gp[p];
        const float g1 = gp[p + 16];
        float* op = out + (((size_t)b * NC + co) * 64 + h) * 64;
        op[wpix]      = (acc0[r] * iv + ad) * g0;
        op[wpix + 16] = (acc1[r] * iv + ad) * g1;
    }
}

// ---------------------------------------------------------------------------
extern "C" void kernel_launch(void* const* d_in, const int* in_sizes, int n_in,
                              void* d_out, int out_size, void* d_ws, size_t ws_size,
                              hipStream_t stream) {
    const float* img  = (const float*)d_in[0];   // (8,256,64,64)
    const float* text = (const float*)d_in[1];   // (8,80,256)
    const float* pw   = (const float*)d_in[2];   // (256,256,1,1)
    const float* pb   = (const float*)d_in[3];   // (256,)
    const float* ab   = (const float*)d_in[4];   // (8,)
    const float* sc   = (const float*)d_in[5];   // (1,8,1,1)
    const float* cwp  = (const float*)d_in[6];   // (256,256,3,3)
    const float* bg   = (const float*)d_in[7];   // bn_gamma
    const float* bb   = (const float*)d_in[8];   // bn_beta
    const float* bm   = (const float*)d_in[9];   // bn_mean
    const float* bv   = (const float*)d_in[10];  // bn_var

    float* ifeat = (float*)d_ws;                  // 8*256*4096 f32 = 33.5 MB
    float* gate  = ifeat + (size_t)NB * NC * NHW; // 8*8*4096 f32 = 1 MB
    float* outp  = (float*)d_out;

    proj_gemm   <<<dim3(64, 4, 8), 128, 0, stream>>>(img, pw, pb, ifeat);
    attn_gate   <<<dim3(64, 8, 8), 128, 0, stream>>>(ifeat, text, ab, sc, gate);
    conv_bn_gate<<<dim3(64, 4, 8), 256, 0, stream>>>(img, cwp, bg, bb, bm, bv, gate, outp);
}
